// AxialSpaceTimeTransformer_48619029791149
// MI455X (gfx1250) — matrix operand order repaired and checked by reading, not compiled, not run on hardware
//
#include <hip/hip_runtime.h>

typedef __bf16 bhalf;
typedef __attribute__((ext_vector_type(16))) __bf16 v16bf;
typedef __attribute__((ext_vector_type(8)))  __bf16 v8bf;
typedef __attribute__((ext_vector_type(8)))  float  v8f;

#define ROWS   8224
#define NDEPTH 8
#define DIMC   512
#define SLEN   257
#define TLEN   16
#define NQH    16
#define NKH    8
#define HDIM   64
#define DIP    1376     // 1365 padded to mult of 32
#define NFF    2752     // 2*DIP

// per-layer bf16 weight blob (element offsets)
#define OFF_QKV  0
#define SZ_QKV   (2048*512)
#define OFF_WO   (OFF_QKV + SZ_QKV)
#define SZ_WO    (512*1024)
#define OFF_WIN  (OFF_WO + SZ_WO)
#define SZ_WIN   (NFF*512)
#define OFF_WOUT (OFF_WIN + SZ_WIN)
#define SZ_WOUT  (512*DIP)
#define WBLOB    (OFF_WOUT + SZ_WOUT)      // 3,686,400 bf16 / layer

#define JMAX 288

// ---------------- helpers ----------------
__device__ inline v8f fzero8() {
  v8f z = {0.f,0.f,0.f,0.f,0.f,0.f,0.f,0.f};
  return z;
}
__device__ inline v8bf bzero8() {
  union { v8bf v; unsigned int u[4]; } z;
  z.u[0]=0u; z.u[1]=0u; z.u[2]=0u; z.u[3]=0u;
  return z.v;
}
__device__ inline v8f wmma_bf16(v16bf a, v16bf b, v8f c) {
  return __builtin_amdgcn_wmma_f32_16x16x32_bf16(false, a, false, b, (short)0, c, false, false);
}

// ---------------- weight conversion ----------------
__global__ void k_copy_f32(const float* __restrict__ s, float* __restrict__ d, int n) {
  for (int i = blockIdx.x*blockDim.x + threadIdx.x; i < n; i += gridDim.x*blockDim.x)
    d[i] = s[i];
}

__global__ void k_cvt_qkv(const float* __restrict__ Wq, const float* __restrict__ Wk,
                          const float* __restrict__ Wv, bhalf* __restrict__ wbf) {
  int l = blockIdx.z;
  const float* wq = Wq + (size_t)l*512*1024;
  const float* wk = Wk + (size_t)l*512*512;
  const float* wv = Wv + (size_t)l*512*512;
  bhalf* d = wbf + (size_t)l*WBLOB + OFF_QKV;
  for (int i = blockIdx.x*blockDim.x + threadIdx.x; i < SZ_QKV; i += gridDim.x*blockDim.x) {
    int n = i >> 9, k = i & 511;
    float v;
    if (n < 1024)      v = wq[(size_t)k*1024 + n];
    else if (n < 1536) v = wk[(size_t)k*512 + (n-1024)];
    else               v = wv[(size_t)k*512 + (n-1536)];
    d[i] = (bhalf)v;
  }
}

__global__ void k_cvt_wo(const float* __restrict__ Wo, bhalf* __restrict__ wbf) {
  int l = blockIdx.z;
  const float* w = Wo + (size_t)l*1024*512;
  bhalf* d = wbf + (size_t)l*WBLOB + OFF_WO;
  for (int i = blockIdx.x*blockDim.x + threadIdx.x; i < SZ_WO; i += gridDim.x*blockDim.x) {
    int n = i >> 10, k = i & 1023;
    d[i] = (bhalf)w[(size_t)k*512 + n];
  }
}

__global__ void k_cvt_win(const float* __restrict__ Win, bhalf* __restrict__ wbf) {
  int l = blockIdx.z;
  const float* w = Win + (size_t)l*512*2730;
  bhalf* d = wbf + (size_t)l*WBLOB + OFF_WIN;
  for (int i = blockIdx.x*blockDim.x + threadIdx.x; i < SZ_WIN; i += gridDim.x*blockDim.x) {
    int n = i >> 9, k = i & 511;
    float v = 0.f;
    if (n < 1365)                   v = w[(size_t)k*2730 + n];
    else if (n >= 1376 && n < 2741) v = w[(size_t)k*2730 + 1365 + (n-1376)];
    d[i] = (bhalf)v;
  }
}

__global__ void k_cvt_wout(const float* __restrict__ Wout, bhalf* __restrict__ wbf) {
  int l = blockIdx.z;
  const float* w = Wout + (size_t)l*1365*512;
  bhalf* d = wbf + (size_t)l*WBLOB + OFF_WOUT;
  for (int i = blockIdx.x*blockDim.x + threadIdx.x; i < SZ_WOUT; i += gridDim.x*blockDim.x) {
    int n = i / DIP, k = i % DIP;
    float v = (k < 1365) ? w[(size_t)k*512 + n] : 0.f;
    d[i] = (bhalf)v;
  }
}

__global__ void k_cvt_bin(const float* __restrict__ bin, float* __restrict__ binp) {
  int l = blockIdx.z;
  const float* b = bin + (size_t)l*2730;
  float* d = binp + (size_t)l*NFF;
  for (int n = blockIdx.x*blockDim.x + threadIdx.x; n < NFF; n += gridDim.x*blockDim.x) {
    float v = 0.f;
    if (n < 1365)                   v = b[n];
    else if (n >= 1376 && n < 2741) v = b[1365 + (n-1376)];
    d[n] = v;
  }
}

// ---------------- RMSNorm ----------------
__global__ __launch_bounds__(128)
void k_rmsnorm(const float* __restrict__ x, const float* __restrict__ w,
               bhalf* __restrict__ ob, float* __restrict__ of) {
  int row = blockIdx.x;
  const float* xr = x + (size_t)row*DIMC;
  int tid = threadIdx.x;
  float4 v = *(const float4*)(xr + tid*4);
  float ss = v.x*v.x + v.y*v.y + v.z*v.z + v.w*v.w;
  for (int m = 16; m >= 1; m >>= 1) ss += __shfl_xor(ss, m, 32);
  __shared__ float wsm[4];
  if ((tid & 31) == 0) wsm[tid >> 5] = ss;
  __syncthreads();
  float tot = wsm[0] + wsm[1] + wsm[2] + wsm[3];
  float r = rsqrtf(tot * (1.f/DIMC) + 1.1920929e-07f);
  const float* wr = w + tid*4;
  float o0 = v.x*r*wr[0], o1 = v.y*r*wr[1], o2 = v.z*r*wr[2], o3 = v.w*r*wr[3];
  if (ob) {
    bhalf* p = ob + (size_t)row*DIMC + tid*4;
    p[0]=(bhalf)o0; p[1]=(bhalf)o1; p[2]=(bhalf)o2; p[3]=(bhalf)o3;
  } else {
    *(float4*)(of + (size_t)row*DIMC + tid*4) = make_float4(o0,o1,o2,o3);
  }
}

// ---------------- bf16 WMMA GEMM: C[MxN] = A[MxK] * Bt[NxK]^T ----------------
#define FBIAS 1
#define FRES  2
#define FBF   4

__global__ __launch_bounds__(256)
void k_gemm(const bhalf* __restrict__ A, const bhalf* __restrict__ Bt,
            float* __restrict__ dstf, bhalf* __restrict__ dstb,
            const float* __restrict__ bias,
            int M, int N, int K, int flags) {
  __shared__ bhalf sA[128*40];
  __shared__ bhalf sB[64*40];
  int tid  = threadIdx.x;
  int mBase = blockIdx.x * 128;
  int nBase = blockIdx.y * 64;
  int wave = tid >> 5, lane = tid & 31;
  int wm = wave & 3, wn = wave >> 2;
  int lr = lane & 15, hi = lane >> 4;

  v8f acc[2][2];
  for (int mt = 0; mt < 2; ++mt) for (int nt = 0; nt < 2; ++nt) acc[mt][nt] = fzero8();

  for (int k0 = 0; k0 < K; k0 += 32) {
    __syncthreads();
    // stage A tile 128x32 (2 x 16B per thread)
    for (int i = 0; i < 2; ++i) {
      int idx = tid + i*256;
      int r = idx >> 2, cs = (idx & 3) * 8;
      int gr = mBase + r;
      v8bf val = bzero8();
      if (gr < M) val = *(const v8bf*)(A + (size_t)gr*K + k0 + cs);
      *(v8bf*)(sA + r*40 + cs) = val;
    }
    // stage B tile 64x32 (1 x 16B per thread)
    {
      int r = tid >> 2, cs = (tid & 3) * 8;
      const bhalf* src = Bt + (size_t)(nBase + r)*K + k0 + cs;
      *(v8bf*)(sB + r*40 + cs) = *(const v8bf*)src;
      if (k0 + 32 < K) __builtin_prefetch(src + 32, 0, 1);   // global_prefetch_b8
    }
    __syncthreads();
    // fragments per documented ISA layouts
    v16bf af[2], bfr[2];
    for (int mt = 0; mt < 2; ++mt) {
      const bhalf* p = sA + (wm*32 + mt*16 + lr)*40;
      union { v16bf v; v8bf h2[2]; } u;
      u.h2[0] = *(const v8bf*)(p + hi*8);
      u.h2[1] = *(const v8bf*)(p + 16 + hi*8);
      af[mt] = u.v;
    }
    for (int nt = 0; nt < 2; ++nt) {
      const bhalf* p = sB + (wn*32 + nt*16 + lr)*40 + hi*8;
      union { v16bf v; v8bf h2[2]; } u;
      u.h2[0] = *(const v8bf*)(p);
      u.h2[1] = *(const v8bf*)(p + 16);
      bfr[nt] = u.v;
    }
    for (int mt = 0; mt < 2; ++mt)
      for (int nt = 0; nt < 2; ++nt)
        acc[mt][nt] = wmma_bf16(af[mt], bfr[nt], acc[mt][nt]);
  }

  for (int mt = 0; mt < 2; ++mt)
    for (int nt = 0; nt < 2; ++nt) {
      int col = nBase + wn*32 + nt*16 + lr;
      for (int r = 0; r < 8; ++r) {
        int row = mBase + wm*32 + mt*16 + r + 8*hi;
        if (row >= M) continue;
        float v = acc[mt][nt][r];
        if (flags & FBIAS) v += bias[col];
        size_t o = (size_t)row*N + col;
        if (flags & FBF) {
          dstb[o] = (bhalf)v;
        } else {
          if (flags & FRES) v += dstf[o];
          dstf[o] = v;
        }
      }
    }
}

// ---------------- q/k post: l2norm * (gamma+1)*sqrt(DH), optional rotary ----------------
__global__ __launch_bounds__(256)
void k_qkpost(bhalf* __restrict__ qkv, const float* __restrict__ qg,
              const float* __restrict__ kg, int use_rot) {
  int row  = blockIdx.y;
  int slot = blockIdx.x*8 + (threadIdx.x >> 5);   // 0..23
  int lane = threadIdx.x & 31;
  int isq  = slot < NQH;
  int h    = isq ? slot : slot - NQH;
  bhalf* p = qkv + (size_t)row*2048 + (isq ? h*64 : 1024 + h*64);
  float v0 = (float)p[lane], v1 = (float)p[lane+32];
  float ss = v0*v0 + v1*v1;
  for (int m = 16; m >= 1; m >>= 1) ss += __shfl_xor(ss, m, 32);
  float inv = 1.f / fmaxf(sqrtf(ss), 1e-12f);
  const float* gm = (isq ? qg : kg) + h*64;
  v0 *= (gm[lane]      + 1.f) * 8.f * inv;
  v1 *= (gm[lane + 32] + 1.f) * 8.f * inv;
  if (use_rot) {
    int t = (row / SLEN) & (TLEN-1);
    float ang = (float)t * __powf(10000.f, -(float)lane * (1.f/32.f));
    float c = __cosf(ang), s = __sinf(ang);
    float n0 = v0*c - v1*s, n1 = v1*c + v0*s;
    v0 = n0; v1 = n1;
  }
  p[lane] = (bhalf)v0; p[lane+32] = (bhalf)v1;
}

// ---------------- build V^T [G][KH][64][Jpad] (zero padded) ----------------
__global__ void k_vt(const bhalf* __restrict__ qkv, bhalf* __restrict__ vT,
                     int n, int Jpad, int mode) {
  int g = blockIdx.x, h = blockIdx.y;
  int base, stride;
  if (mode == 0) { base = g*SLEN; stride = 1; }
  else           { base = (g/SLEN)*(TLEN*SLEN) + (g%SLEN); stride = SLEN; }
  bhalf* dst = vT + ((size_t)(g*NKH + h)*64)*Jpad;
  int tot = 64 * Jpad;
  for (int e = threadIdx.x; e < tot; e += blockDim.x) {
    int d = e / Jpad, j = e % Jpad;
    bhalf v = (bhalf)0.f;
    if (j < n) v = qkv[(size_t)(base + j*stride)*2048 + 1536 + h*64 + d];
    dst[e] = v;
  }
}

// ---------------- attention (both axial modes) ----------------
__global__ __launch_bounds__(128)
void k_attn(const bhalf* __restrict__ qkv, const bhalf* __restrict__ vT,
            bhalf* __restrict__ o, int n, int nck, int Jpad, int mode) {
  __shared__ float ssim[16*JMAX];
  __shared__ bhalf sp[16*JMAX];
  __shared__ float part[16*8];
  __shared__ float rowstat[16];

  int qc = blockIdx.x, h = blockIdx.y, g = blockIdx.z;
  int hkv = h >> 1;
  int tid = threadIdx.x, wave = tid >> 5, lane = tid & 31;
  int lr = lane & 15, hi = lane >> 4;
  int base, stride;
  if (mode == 0) { base = g*SLEN; stride = 1; }
  else           { base = (g/SLEN)*(TLEN*SLEN) + (g%SLEN); stride = SLEN; }

  // Q fragments (16 queries x 64 feat -> 2 k-chunks), rows clamped for OOB queries
  v16bf aq[2];
  {
    int qi = qc*16 + lr; if (qi > n-1) qi = n-1;
    const bhalf* qrow = qkv + (size_t)(base + qi*stride)*2048 + h*64;
    for (int kk = 0; kk < 2; ++kk) {
      union { v16bf v; v8bf h2[2]; } u;
      u.h2[0] = *(const v8bf*)(qrow + kk*32 + hi*8);
      u.h2[1] = *(const v8bf*)(qrow + kk*32 + 16 + hi*8);
      aq[kk] = u.v;
    }
  }

  // S = softclamp(Q K^T / sqrt(d)) with masks, tiled 16 keys per wave
  for (int c = wave; c < nck; c += 4) {
    int j = c*16 + lr;                       // global key index == column
    bool valid = (j < n);
    int jj = valid ? j : 0;
    const bhalf* krow = qkv + (size_t)(base + jj*stride)*2048 + 1024 + hkv*64;
    v16bf bk[2];
    for (int kk = 0; kk < 2; ++kk) {
      union { v16bf v; v8bf h2[2]; } u;
      if (valid) {
        u.h2[0] = *(const v8bf*)(krow + kk*32 + hi*8);
        u.h2[1] = *(const v8bf*)(krow + kk*32 + 16 + hi*8);
      } else { u.h2[0] = bzero8(); u.h2[1] = bzero8(); }
      bk[kk] = u.v;
    }
    v8f s = fzero8();
    s = wmma_bf16(aq[0], bk[0], s);
    s = wmma_bf16(aq[1], bk[1], s);
    for (int r = 0; r < 8; ++r) {
      int qgi = qc*16 + r + 8*hi;
      float v = s[r] * 0.125f;               // 1/sqrt(64)
      v = tanhf(v * 0.02f) * 50.f;           // softclamp
      bool msk = (j >= n) || (qgi >= n);
      if (mode == 0) msk = msk || (j == n-1 && qgi != n-1);  // special-token mask
      else           msk = msk || (j > qgi);                 // causal
      if (msk) v = -3.0e38f;
      ssim[(r + 8*hi)*JMAX + c*16 + lr] = v;
    }
  }
  __syncthreads();

  // softmax over key dim (rows of 16-query chunk)
  int row = tid >> 3, seg = tid & 7;
  int lim = nck * 16;
  float m = -3.0e38f;
  for (int j = seg; j < lim; j += 8) m = fmaxf(m, ssim[row*JMAX + j]);
  part[row*8 + seg] = m;
  __syncthreads();
  if (seg == 0) {
    float mm = part[row*8];
    for (int i = 1; i < 8; ++i) mm = fmaxf(mm, part[row*8 + i]);
    rowstat[row] = mm;
  }
  __syncthreads();
  float rm = rowstat[row];
  float sum = 0.f;
  for (int j = seg; j < lim; j += 8) {
    float e = __expf(ssim[row*JMAX + j] - rm);
    ssim[row*JMAX + j] = e;
    sum += e;
  }
  part[row*8 + seg] = sum;
  __syncthreads();
  if (seg == 0) {
    float t = 0.f;
    for (int i = 0; i < 8; ++i) t += part[row*8 + i];
    rowstat[row] = t;
  }
  __syncthreads();
  float rd = 1.f / rowstat[row];
  for (int j = seg; j < Jpad; j += 8)
    sp[row*JMAX + j] = (bhalf)((j < lim) ? ssim[row*JMAX + j] * rd : 0.f);
  __syncthreads();

  // O = P * V   (wave w -> feature tile w)
  v8f accO = fzero8();
  const bhalf* vbase = vT + ((size_t)(g*NKH + hkv)*64 + wave*16 + lr)*Jpad;
  const bhalf* prow  = sp + lr*JMAX;
  for (int kk = 0; kk < Jpad/32; ++kk) {
    union { v16bf v; v8bf h2[2]; } ua;
    ua.h2[0] = *(const v8bf*)(prow + kk*32 + hi*8);
    ua.h2[1] = *(const v8bf*)(prow + kk*32 + 16 + hi*8);
    union { v16bf v; v8bf h2[2]; } ub;
    ub.h2[0] = *(const v8bf*)(vbase + kk*32 + hi*8);
    ub.h2[1] = *(const v8bf*)(vbase + kk*32 + 16 + hi*8);
    accO = wmma_bf16(ua.v, ub.v, accO);
  }
  for (int r = 0; r < 8; ++r) {
    int qgi = qc*16 + r + 8*hi;
    if (qgi < n)
      o[(size_t)(base + qgi*stride)*1024 + h*64 + wave*16 + lr] = (bhalf)accO[r];
  }
}

// ---------------- gated-GELU activation ----------------
__global__ void k_act(const bhalf* __restrict__ hff, bhalf* __restrict__ u) {
  size_t total = (size_t)ROWS * DIP;
  size_t i = (size_t)blockIdx.x*blockDim.x + threadIdx.x;
  if (i >= total) return;
  size_t m = i / DIP, j = i % DIP;
  float a = (float)hff[m*NFF + j];
  float g = (float)hff[m*NFF + DIP + j];
  float gl = 0.5f * g * (1.f + erff(g * 0.70710678118654752f));
  u[i] = (bhalf)(a * gl);
}

// ---------------- workspace layout (bytes, all 256-aligned) ----------------
static constexpr size_t OWB   = 0;
static constexpr size_t SZWB  = (size_t)WBLOB * NDEPTH * 2;     // 58,982,400
static constexpr size_t OBIN  = OWB + SZWB;
static constexpr size_t SZBIN = (size_t)NDEPTH * NFF * 4;       // 88,064
static constexpr size_t OX    = OBIN + SZBIN;
static constexpr size_t SZX   = (size_t)ROWS * DIMC * 4;        // 16,842,752
static constexpr size_t OH    = OX + SZX;
static constexpr size_t SZH   = (size_t)ROWS * DIMC * 2;        // 8,421,376
static constexpr size_t OSCR  = OH + SZH;
static constexpr size_t OQKV  = OSCR;
static constexpr size_t SZQKV = (size_t)ROWS * 2048 * 2;        // 33,685,504
static constexpr size_t OVT   = OQKV + SZQKV;
static constexpr size_t SZVT  = (size_t)514 * NKH * HDIM * 32 * 2;  // 16,842,752 (max of both modes)
static constexpr size_t OO    = OVT + SZVT;
static constexpr size_t SZO   = (size_t)ROWS * 1024 * 2;        // 16,842,752
static constexpr size_t OHFF  = OSCR;                           // aliases attention scratch
static constexpr size_t SZHFF = (size_t)ROWS * NFF * 2;         // 45,264,896
static constexpr size_t OU    = OHFF + SZHFF;
static constexpr size_t SZU   = (size_t)ROWS * DIP * 2;         // 22,632,448
static constexpr size_t SZATT = SZQKV + SZVT + SZO;
static constexpr size_t SZFF  = SZHFF + SZU;
static constexpr size_t WTOT  = OSCR + (SZATT > SZFF ? SZATT : SZFF);

extern "C" void kernel_launch(void* const* d_in, const int* in_sizes, int n_in,
                              void* d_out, int out_size, void* d_ws, size_t ws_size,
                              hipStream_t stream) {
  (void)in_sizes; (void)n_in; (void)out_size;
  if (ws_size < WTOT) return;

  const float* tokens      = (const float*)d_in[0];
  const float* attn_norm_w = (const float*)d_in[1];
  const float* Wq          = (const float*)d_in[2];
  const float* Wk          = (const float*)d_in[3];
  const float* Wv          = (const float*)d_in[4];
  const float* q_gamma     = (const float*)d_in[5];
  const float* k_gamma     = (const float*)d_in[6];
  const float* Wo          = (const float*)d_in[7];
  const float* ff_norm_w   = (const float*)d_in[8];
  const float* W_in        = (const float*)d_in[9];
  const float* b_in        = (const float*)d_in[10];
  const float* W_out       = (const float*)d_in[11];
  const float* b_out       = (const float*)d_in[12];
  const float* final_w     = (const float*)d_in[13];

  char*  ws   = (char*)d_ws;
  bhalf* wbf  = (bhalf*)(ws + OWB);
  float* binp = (float*)(ws + OBIN);
  float* x    = (float*)(ws + OX);
  bhalf* h    = (bhalf*)(ws + OH);
  bhalf* qkv  = (bhalf*)(ws + OQKV);
  bhalf* vT   = (bhalf*)(ws + OVT);
  bhalf* o    = (bhalf*)(ws + OO);
  bhalf* hff  = (bhalf*)(ws + OHFF);
  bhalf* u    = (bhalf*)(ws + OU);

  // weights -> bf16 (transposed/padded), x <- tokens
  k_cvt_qkv <<<dim3(256,1,NDEPTH), 256, 0, stream>>>(Wq, Wk, Wv, wbf);
  k_cvt_wo  <<<dim3(256,1,NDEPTH), 256, 0, stream>>>(Wo, wbf);
  k_cvt_win <<<dim3(256,1,NDEPTH), 256, 0, stream>>>(W_in, wbf);
  k_cvt_wout<<<dim3(256,1,NDEPTH), 256, 0, stream>>>(W_out, wbf);
  k_cvt_bin <<<dim3(11,1,NDEPTH),  256, 0, stream>>>(b_in, binp);
  k_copy_f32<<<4096, 256, 0, stream>>>(tokens, x, ROWS*DIMC);

  const int MT = (ROWS + 127) / 128;  // 65
  for (int l = 0; l < NDEPTH; ++l) {
    const bhalf* wl = wbf + (size_t)l*WBLOB;
    int is_time = ((l+1) % 4) == 0;
    int n    = is_time ? TLEN : SLEN;
    int nck  = is_time ? 1 : 17;
    int Jpad = is_time ? 32 : 288;
    int mode = is_time ? 1 : 0;
    int G    = is_time ? 514 : 32;

    // ---- attention ----
    k_rmsnorm<<<ROWS, 128, 0, stream>>>(x, attn_norm_w + l*DIMC, h, nullptr);
    k_gemm<<<dim3(MT, 2048/64), 256, 0, stream>>>(h, wl + OFF_QKV,
        nullptr, qkv, nullptr, ROWS, 2048, 512, FBF);
    k_qkpost<<<dim3(3, ROWS), 256, 0, stream>>>(qkv,
        q_gamma + l*NQH*HDIM, k_gamma + l*NKH*HDIM, is_time);
    k_vt<<<dim3(G, NKH), 256, 0, stream>>>(qkv, vT, n, Jpad, mode);
    k_attn<<<dim3(nck, NQH, G), 128, 0, stream>>>(qkv, vT, o, n, nck, Jpad, mode);
    k_gemm<<<dim3(MT, 512/64), 256, 0, stream>>>(o, wl + OFF_WO,
        x, nullptr, nullptr, ROWS, 512, 1024, FRES);

    // ---- feed-forward ----
    k_rmsnorm<<<ROWS, 128, 0, stream>>>(x, ff_norm_w + l*DIMC, h, nullptr);
    k_gemm<<<dim3(MT, NFF/64), 256, 0, stream>>>(h, wl + OFF_WIN,
        nullptr, hff, binp + (size_t)l*NFF, ROWS, NFF, 512, FBF | FBIAS);
    {
      size_t tot = (size_t)ROWS * DIP;
      int blocks = (int)((tot + 255) / 256);
      k_act<<<blocks, 256, 0, stream>>>(hff, u);
    }
    k_gemm<<<dim3(MT, 512/64), 256, 0, stream>>>(u, wl + OFF_WOUT,
        x, nullptr, b_out + l*DIMC, ROWS, 512, DIP, FRES | FBIAS);
  }

  // final RMSNorm -> d_out (f32)
  k_rmsnorm<<<ROWS, 128, 0, stream>>>(x, final_w, nullptr, (float*)d_out);
}
